// PMM_1065151889576
// MI455X (gfx1250) — matrix operand order repaired and checked by reading, not compiled, not run on hardware
//
#include <hip/hip_runtime.h>
#include <hip/hip_bf16.h>
#include <stdint.h>

// ---------------------------------------------------------------------------
// Types for CDNA5 WMMA / TDM builtins
// ---------------------------------------------------------------------------
typedef __bf16 bf16_t;
typedef __bf16 v16bf __attribute__((ext_vector_type(16)));
typedef float  v8f   __attribute__((ext_vector_type(8)));
typedef unsigned int u32x4 __attribute__((ext_vector_type(4)));
typedef int    i32x8 __attribute__((ext_vector_type(8)));
typedef int    i32x4 __attribute__((ext_vector_type(4)));

#define LL   4096     // H*W
#define CC   512
#define DM   128
#define DI   256
#define DRR  8
#define DSS  16
#define NB   8        // 4 * B
#define MROWS (NB * LL)   // 32768 token rows through the shared mamba

// ---------------------------------------------------------------------------
// Stage a (rows x cols) bf16 weight tile from global into LDS.
// Preferred path: Tensor Data Mover (tensor_load_to_lds, 6-arg toolchain form)
// with a D# built per CDNA5 ISA (2D tile, data_size=2B),
// waited with s_wait_tensorcnt. Fallback: cooperative copy (host pass /
// toolchains without the builtin).
// ---------------------------------------------------------------------------
__device__ inline void lds_load_w_tile(bf16_t* sW, const bf16_t* gW,
                                       int rows, int cols, int stride) {
#if defined(__AMDGCN__) && __has_builtin(__builtin_amdgcn_tensor_load_to_lds) && __has_builtin(__builtin_amdgcn_s_wait_tensorcnt)
  if ((threadIdx.x >> 5) == 0) {  // one wave issues; EXEC is ignored by TDM
    unsigned long long ga = (unsigned long long)(uintptr_t)gW;   // global byte addr
    unsigned lds = (unsigned)(uintptr_t)sW;                      // LDS byte offset (flat low 32)
    u32x4 g0;
    g0[0] = 1u;                                  // count=1 (valid descriptor)
    g0[1] = lds;                                 // lds_addr
    g0[2] = (unsigned)ga;                        // global_addr[31:0]
    g0[3] = (unsigned)(ga >> 32) | 0x80000000u;  // global_addr[56:32] | type=2<<30
    i32x8 g1;
    g1[0] = 1 << 16;                                             // data_size=1 (2 bytes)
    g1[1] = (cols & 0xFFFF) << 16;                               // tensor_dim0 lo16
    g1[2] = ((cols >> 16) & 0xFFFF) | ((rows & 0xFFFF) << 16);   // tdim0 hi | tensor_dim1 lo
    g1[3] = ((rows >> 16) & 0xFFFF) | ((cols & 0xFFFF) << 16);   // tdim1 hi | tile_dim0
    g1[4] = (rows & 0xFFFF);                                     // tile_dim1 (tile_dim2=0)
    g1[5] = stride;                                              // tensor_dim0_stride lo32
    g1[6] = 0;                                                   // stride hi | tdim1_stride lo
    g1[7] = 0;
    i32x4 z4; z4[0] = 0; z4[1] = 0; z4[2] = 0; z4[3] = 0;        // groups 2/3: NULL (2D tile)
    i32x8 z8;
    z8[0] = 0; z8[1] = 0; z8[2] = 0; z8[3] = 0;
    z8[4] = 0; z8[5] = 0; z8[6] = 0; z8[7] = 0;
    __builtin_amdgcn_tensor_load_to_lds(g0, g1, z4, z4, z8, 0);  // 6-arg toolchain form
    __builtin_amdgcn_s_wait_tensorcnt(0);
  }
#else
  for (int i = threadIdx.x; i < rows * cols; i += blockDim.x) {
    int rr = i / cols, cc2 = i - rr * cols;
    sW[i] = gW[rr * stride + cc2];
  }
#endif
  __syncthreads();
}

// ---------------------------------------------------------------------------
// One wave computes a 16x16 f32 tile:  C = A(16xK, row-major bf16, stride lda)
//                                         * W(16 rows of (N,K) weights, i.e. B^T)
// K is compile-time -> fully unrolled chain of v_wmma_f32_16x16x32_bf16.
// A fragment layout per ISA 7.12.2 (16-bit A 16x32); B uses the same
// k order as A (kb = ka: lanes 0-15 K=0..7,16..23; lanes 16-31 K=8..15,24..31).
// ---------------------------------------------------------------------------
template <int K>
__device__ inline v8f wmma_tile(const bf16_t* __restrict__ A, int lda,
                                const bf16_t* __restrict__ Wl, int ldw) {
  const int lane = threadIdx.x & 31;
  const int half = lane >> 4;
  const int r    = lane & 15;
  v8f acc = {};
  #pragma unroll
  for (int k0 = 0; k0 < K; k0 += 32) {
    v16bf a, b;
    #pragma unroll
    for (int e = 0; e < 16; ++e) {
      const int p = e & 1, v = e >> 1;
      const int ka = ((v < 4) ? (8 * half + 2 * v) : (16 + 8 * half + 2 * (v - 4))) + p;
      a[e] = A[r * lda + k0 + ka];
      const int kb = ka;
      b[e] = Wl[r * ldw + k0 + kb];
    }
    acc = __builtin_amdgcn_wmma_f32_16x16x32_bf16(false, a, false, b, (short)0, acc,
                                                  false, false);
  }
  return acc;
}

// ---------------------------------------------------------------------------
// Utility: f32 -> bf16 with optional zero row padding (dst has padRows rows)
// ---------------------------------------------------------------------------
__global__ void k_convert_bf16(const float* __restrict__ src, bf16_t* __restrict__ dst,
                               int rows, int cols, int padRows) {
  int t = blockIdx.x * 256 + threadIdx.x;
  if (t >= padRows * cols) return;
  int rr = t / cols;
  dst[t] = (rr < rows) ? (bf16_t)src[t] : (bf16_t)0.0f;
}

// ---------------------------------------------------------------------------
// LayerNorm over channels: x (B,C,H,W) -> xn (B,L,C) f32 + bf16
// one block per (b,l), 256 threads handle 2 channels each
// ---------------------------------------------------------------------------
__global__ void k_layernorm(const float* __restrict__ x, const float* __restrict__ g,
                            const float* __restrict__ bta, float* __restrict__ xn,
                            bf16_t* __restrict__ xnbf) {
  __shared__ float red[256];
  const int bl = blockIdx.x;             // 0 .. B*L-1
  const int b = bl >> 12, l = bl & (LL - 1);
  const float* xp = x + (size_t)b * CC * LL + l;
  const int c0 = threadIdx.x * 2;
  const float v0 = xp[(size_t)c0 * LL];
  const float v1 = xp[(size_t)(c0 + 1) * LL];
  red[threadIdx.x] = v0 + v1;
  __syncthreads();
  for (int st = 128; st; st >>= 1) {
    if (threadIdx.x < st) red[threadIdx.x] += red[threadIdx.x + st];
    __syncthreads();
  }
  const float mean = red[0] * (1.0f / CC);
  __syncthreads();
  red[threadIdx.x] = (v0 - mean) * (v0 - mean) + (v1 - mean) * (v1 - mean);
  __syncthreads();
  for (int st = 128; st; st >>= 1) {
    if (threadIdx.x < st) red[threadIdx.x] += red[threadIdx.x + st];
    __syncthreads();
  }
  const float inv = rsqrtf(red[0] * (1.0f / CC) + 1e-5f);
  const size_t ob = ((size_t)b * LL + l) * CC;
  const float o0 = (v0 - mean) * inv * g[c0] + bta[c0];
  const float o1 = (v1 - mean) * inv * g[c0 + 1] + bta[c0 + 1];
  xn[ob + c0] = o0;     xn[ob + c0 + 1] = o1;
  xnbf[ob + c0] = (bf16_t)o0;  xnbf[ob + c0 + 1] = (bf16_t)o1;
}

// ---------------------------------------------------------------------------
// in_proj: xz[m,512] = xc[m,:128] @ in_proj_w^T   (m = n*L+l, n=(chunk,b))
// grid (M_tiles/8=256, N_tiles=32), 8 waves/block, W tile staged via TDM
// ---------------------------------------------------------------------------
__global__ void k_gemm_in_proj(const bf16_t* __restrict__ xnbf,
                               const bf16_t* __restrict__ wbf,
                               float* __restrict__ xz) {
  __shared__ bf16_t sW[16 * 128];
  const int N0 = blockIdx.y * 16;
  lds_load_w_tile(sW, wbf + N0 * 128, 16, 128, 128);
  const int m0 = (blockIdx.x * 8 + (threadIdx.x >> 5)) * 16;
  const int n = m0 >> 12, l0 = m0 & (LL - 1);
  const int b = n & 1, ch = n >> 1;
  const bf16_t* A = xnbf + ((size_t)(b * LL + l0)) * CC + ch * DM;
  v8f acc = wmma_tile<128>(A, CC, sW, 128);
  const int lane = threadIdx.x & 31, half = lane >> 4, r = lane & 15;
  #pragma unroll
  for (int i = 0; i < 8; ++i)
    xz[(size_t)(m0 + i + 8 * half) * 512 + N0 + r] = acc[i];
}

// ---------------------------------------------------------------------------
// causal depthwise conv1d (k=4) + bias + SiLU : xm half of xz -> u (f32+bf16)
// ---------------------------------------------------------------------------
__global__ void k_conv_silu(const float* __restrict__ xz, const float* __restrict__ cw,
                            const float* __restrict__ cb, float* __restrict__ u,
                            bf16_t* __restrict__ ubf) {
  const size_t t = (size_t)blockIdx.x * 256 + threadIdx.x;  // over NB*L*DI
  const int d = (int)(t & (DI - 1));
  const size_t ml = t >> 8;
  const int l = (int)(ml & (LL - 1));
  const int n = (int)(ml >> 12);
  float acc = cb[d];
  #pragma unroll
  for (int j = 0; j < 4; ++j) {
    const int ll = l - 3 + j;
    if (ll >= 0) acc += xz[((size_t)(n * LL + ll)) * 512 + d] * cw[d * 4 + j];
  }
  const float s = acc / (1.0f + __expf(-acc));
  u[t] = s;
  ubf[t] = (bf16_t)s;
}

// ---------------------------------------------------------------------------
// x_proj: x_dbl[m, 0:40] = u[m,:256] @ x_proj_w^T ; W zero-padded to 48 rows
// stored with ldc=48. grid (256, 3)
// ---------------------------------------------------------------------------
__global__ void k_gemm_x_proj(const bf16_t* __restrict__ ubf,
                              const bf16_t* __restrict__ wbf,
                              float* __restrict__ xdbl) {
  __shared__ bf16_t sW[16 * 256];
  const int N0 = blockIdx.y * 16;
  lds_load_w_tile(sW, wbf + N0 * 256, 16, 256, 256);
  const int m0 = (blockIdx.x * 8 + (threadIdx.x >> 5)) * 16;
  const bf16_t* A = ubf + (size_t)m0 * DI;
  v8f acc = wmma_tile<256>(A, DI, sW, 256);
  const int lane = threadIdx.x & 31, half = lane >> 4, r = lane & 15;
  #pragma unroll
  for (int i = 0; i < 8; ++i)
    xdbl[(size_t)(m0 + i + 8 * half) * 48 + N0 + r] = acc[i];
}

// ---------------------------------------------------------------------------
// dt = softplus(x_dbl[:, :8] @ dt_w^T + dt_b)  (K=8, plain VALU)
// ---------------------------------------------------------------------------
__global__ void k_dt(const float* __restrict__ xdbl, const float* __restrict__ dtw,
                     const float* __restrict__ dtb, float* __restrict__ dt) {
  const size_t t = (size_t)blockIdx.x * 256 + threadIdx.x;  // over MROWS*DI
  const int d = (int)(t & (DI - 1));
  const size_t m = t >> 8;
  float acc = dtb[d];
  #pragma unroll
  for (int r = 0; r < DRR; ++r) acc += xdbl[m * 48 + r] * dtw[d * DRR + r];
  dt[t] = (acc > 20.0f) ? acc : log1pf(__expf(acc));
}

// ---------------------------------------------------------------------------
// Selective scan: one lane per (n,d,s); 1024 wave32 waves; y reduced over the
// 16 state lanes with __shfl_xor (width 16 -> ds_swizzle/permlane on wave32).
// ---------------------------------------------------------------------------
__global__ void k_scan(const float* __restrict__ u, const float* __restrict__ dt,
                       const float* __restrict__ xdbl, const float* __restrict__ A_log,
                       const float* __restrict__ Dp, float* __restrict__ y) {
  const int t = blockIdx.x * 256 + threadIdx.x;  // 0 .. 32767
  const int s = t & 15;
  const int d = (t >> 4) & (DI - 1);
  const int n = t >> 12;
  const float Av = -__expf(A_log[d * DSS + s]);
  const float Dd = Dp[d];
  float h = 0.0f;
  for (int l = 0; l < LL; ++l) {
    const size_t m = ((size_t)n << 12) + l;
    const float dtv = dt[m * DI + d];
    const float uv = u[m * DI + d];
    const float Bt = xdbl[m * 48 + 8 + s];
    const float Ct = xdbl[m * 48 + 24 + s];
    if (l + 1 < LL) {
      __builtin_prefetch(&dt[(m + 1) * DI + d], 0, 1);   // global_prefetch_b8
      __builtin_prefetch(&xdbl[(m + 1) * 48 + s], 0, 1);
    }
    h = __expf(dtv * Av) * h + dtv * uv * Bt;
    float part = h * Ct;
    #pragma unroll
    for (int off = 8; off; off >>= 1) part += __shfl_xor(part, off, 16);
    if (s == 0) y[m * DI + d] = part + uv * Dd;
  }
}

// ---------------------------------------------------------------------------
// gate: y *= silu(z)   (z = cols 256..511 of xz), also emit bf16 for out_proj
// ---------------------------------------------------------------------------
__global__ void k_gate(const float* __restrict__ xz, float* __restrict__ y,
                       bf16_t* __restrict__ ybf) {
  const size_t t = (size_t)blockIdx.x * 256 + threadIdx.x;
  const int d = (int)(t & (DI - 1));
  const size_t m = t >> 8;
  const float zv = xz[m * 512 + DI + d];
  const float v = y[t] * (zv / (1.0f + __expf(-zv)));
  y[t] = v;
  ybf[t] = (bf16_t)v;
}

// ---------------------------------------------------------------------------
// out_proj + residual + permute back to (B,C,H,W):
// ybchw[b, ch*128+o, l] = (y[m,:256] @ out_proj_w^T)[o] + xn[b,l,c]
// ---------------------------------------------------------------------------
__global__ void k_gemm_out_proj(const bf16_t* __restrict__ ybf,
                                const bf16_t* __restrict__ wbf,
                                const float* __restrict__ xn,
                                float* __restrict__ ybchw) {
  __shared__ bf16_t sW[16 * 256];
  const int N0 = blockIdx.y * 16;
  lds_load_w_tile(sW, wbf + N0 * 256, 16, 256, 256);
  const int m0 = (blockIdx.x * 8 + (threadIdx.x >> 5)) * 16;
  const int n = m0 >> 12, l0 = m0 & (LL - 1);
  const int b = n & 1, ch = n >> 1;
  const bf16_t* A = ybf + (size_t)m0 * DI;
  v8f acc = wmma_tile<256>(A, DI, sW, 256);
  const int lane = threadIdx.x & 31, half = lane >> 4, r = lane & 15;
  const int c = ch * DM + N0 + r;
  #pragma unroll
  for (int i = 0; i < 8; ++i) {
    const int l = l0 + i + 8 * half;
    ybchw[((size_t)(b * CC + c) << 12) + l] =
        acc[i] + xn[((size_t)(b * LL + l)) * CC + c];
  }
}

// ---------------------------------------------------------------------------
// BatchNorm2d (training stats) over (B,H,W) per channel
// ---------------------------------------------------------------------------
__global__ void k_bn_stats(const float* __restrict__ ybchw, float* __restrict__ stats) {
  __shared__ float s1[256], s2[256];
  const int c = blockIdx.x;
  float a = 0.0f, b2 = 0.0f;
  for (int i = threadIdx.x; i < 2 * LL; i += 256) {
    const int b = i >> 12, l = i & (LL - 1);
    const float v = ybchw[((size_t)(b * CC + c) << 12) + l];
    a += v; b2 += v * v;
  }
  s1[threadIdx.x] = a; s2[threadIdx.x] = b2;
  __syncthreads();
  for (int st = 128; st; st >>= 1) {
    if (threadIdx.x < st) { s1[threadIdx.x] += s1[threadIdx.x + st];
                            s2[threadIdx.x] += s2[threadIdx.x + st]; }
    __syncthreads();
  }
  if (threadIdx.x == 0) {
    const float mean = s1[0] * (1.0f / (2 * LL));
    const float var = s2[0] * (1.0f / (2 * LL)) - mean * mean;
    stats[c] = mean;
    stats[CC + c] = rsqrtf(var + 1e-5f);
  }
}

__global__ void k_bn_apply(const float* __restrict__ ybchw, const float* __restrict__ stats,
                           const float* __restrict__ g, const float* __restrict__ bb,
                           float* __restrict__ out_mamba, bf16_t* __restrict__ xmb) {
  const size_t t = (size_t)blockIdx.x * 256 + threadIdx.x;  // over B*C*L
  const int l = (int)(t & (LL - 1));
  const int c = (int)((t >> 12) & (CC - 1));
  const int b = (int)(t >> 21);
  const float v = (ybchw[t] - stats[c]) * stats[CC + c] * g[c] + bb[c];
  out_mamba[t] = v;                                        // (B,C,H,W) flat == t
  xmb[((size_t)(b * LL + l)) * CC + c] = (bf16_t)v;        // (B,L,C) for pro1
}

// ---------------------------------------------------------------------------
// pro1: f[b,o,l] = x_mamba[b,:,l] . pro1_w[o,:] + pro1_b[o]   (K=512)
// ---------------------------------------------------------------------------
__global__ void k_gemm_pro1(const bf16_t* __restrict__ xmb, const bf16_t* __restrict__ wbf,
                            const float* __restrict__ bias, float* __restrict__ f) {
  __shared__ bf16_t sW[16 * 512];
  const int N0 = blockIdx.y * 16;
  lds_load_w_tile(sW, wbf + N0 * 512, 16, 512, 512);
  const int m0 = (blockIdx.x * 8 + (threadIdx.x >> 5)) * 16;   // m = b*L + l
  const int b = m0 >> 12, l0 = m0 & (LL - 1);
  const bf16_t* A = xmb + (size_t)m0 * CC;
  v8f acc = wmma_tile<512>(A, CC, sW, 512);
  const int lane = threadIdx.x & 31, half = lane >> 4, r = lane & 15;
  const int o = N0 + r;
  #pragma unroll
  for (int i = 0; i < 8; ++i) {
    const int l = l0 + i + 8 * half;
    f[((size_t)(b * CC + o) << 12) + l] = acc[i] + bias[o];
  }
}

// ---------------------------------------------------------------------------
// depthwise 7x7 conv (pad 3) + bias + exact GELU -> bf16 (B,L,C) for pro2
// ---------------------------------------------------------------------------
__global__ void k_dwconv_gelu(const float* __restrict__ f, const float* __restrict__ w,
                              const float* __restrict__ bias, bf16_t* __restrict__ f2bf) {
  const size_t t = (size_t)blockIdx.x * 256 + threadIdx.x;  // over B*C*L
  const int l = (int)(t & (LL - 1));
  const int c = (int)((t >> 12) & (CC - 1));
  const int b = (int)(t >> 21);
  const int h = l >> 6, wc = l & 63;
  float acc = bias[c];
  const float* fp = f + ((size_t)(b * CC + c) << 12);
  #pragma unroll
  for (int ky = 0; ky < 7; ++ky) {
    const int hh = h + ky - 3;
    if (hh < 0 || hh >= 64) continue;
    #pragma unroll
    for (int kx = 0; kx < 7; ++kx) {
      const int ww = wc + kx - 3;
      if (ww < 0 || ww >= 64) continue;
      acc += fp[hh * 64 + ww] * w[c * 49 + ky * 7 + kx];
    }
  }
  const float ge = 0.5f * acc * (1.0f + erff(acc * 0.70710678118654752f));
  f2bf[((size_t)(b * LL + l)) * CC + c] = (bf16_t)ge;
}

// ---------------------------------------------------------------------------
// pro2: x_ffn[b,o,l] = f[b,:,l] . pro2_w[o,:] + pro2_b[o]  -> d_out (2nd half)
// ---------------------------------------------------------------------------
__global__ void k_gemm_pro2(const bf16_t* __restrict__ f2bf, const bf16_t* __restrict__ wbf,
                            const float* __restrict__ bias, float* __restrict__ out_ffn) {
  __shared__ bf16_t sW[16 * 512];
  const int N0 = blockIdx.y * 16;
  lds_load_w_tile(sW, wbf + N0 * 512, 16, 512, 512);
  const int m0 = (blockIdx.x * 8 + (threadIdx.x >> 5)) * 16;
  const int b = m0 >> 12, l0 = m0 & (LL - 1);
  const bf16_t* A = f2bf + (size_t)m0 * CC;
  v8f acc = wmma_tile<512>(A, CC, sW, 512);
  const int lane = threadIdx.x & 31, half = lane >> 4, r = lane & 15;
  const int o = N0 + r;
  #pragma unroll
  for (int i = 0; i < 8; ++i) {
    const int l = l0 + i + 8 * half;
    out_ffn[((size_t)(b * CC + o) << 12) + l] = acc[i] + bias[o];
  }
}

// ---------------------------------------------------------------------------
// host launcher
// ---------------------------------------------------------------------------
extern "C" void kernel_launch(void* const* d_in, const int* in_sizes, int n_in,
                              void* d_out, int out_size, void* d_ws, size_t ws_size,
                              hipStream_t stream) {
  (void)in_sizes; (void)n_in; (void)out_size; (void)ws_size;

  const float* x         = (const float*)d_in[0];
  const float* ln_g      = (const float*)d_in[1];
  const float* ln_b      = (const float*)d_in[2];
  const float* in_proj_w = (const float*)d_in[3];
  const float* conv_w    = (const float*)d_in[4];
  const float* conv_b    = (const float*)d_in[5];
  const float* x_proj_w  = (const float*)d_in[6];
  const float* dt_w      = (const float*)d_in[7];
  const float* dt_b      = (const float*)d_in[8];
  const float* A_log     = (const float*)d_in[9];
  const float* Dp        = (const float*)d_in[10];
  const float* out_proj_w= (const float*)d_in[11];
  const float* bn_g      = (const float*)d_in[12];
  const float* bn_b      = (const float*)d_in[13];
  const float* pro1_w    = (const float*)d_in[14];
  const float* pro1_b    = (const float*)d_in[15];
  const float* dw_w      = (const float*)d_in[16];
  const float* dw_b      = (const float*)d_in[17];
  const float* pro2_w    = (const float*)d_in[18];
  const float* pro2_b    = (const float*)d_in[19];

  char* ws = (char*)d_ws;
  size_t off = 0;
  auto alloc = [&](size_t bytes) { size_t r = off; off += (bytes + 255) & ~(size_t)255; return r; };

  float*  XN    = (float*) (ws + alloc((size_t)2 * LL * CC * 4));        // (B,L,C)
  bf16_t* XNBF  = (bf16_t*)(ws + alloc((size_t)2 * LL * CC * 2));
  float*  XZ    = (float*) (ws + alloc((size_t)MROWS * 512 * 4));        // xm|z
  float*  U     = (float*) (ws + alloc((size_t)MROWS * DI * 4));
  bf16_t* UBF   = (bf16_t*)(ws + alloc((size_t)MROWS * DI * 2));         // reused as YBF
  float*  XDBL  = (float*) (ws + alloc((size_t)MROWS * 48 * 4));
  float*  DT    = (float*) (ws + alloc((size_t)MROWS * DI * 4));
  float*  Y     = (float*) (ws + alloc((size_t)MROWS * DI * 4));
  float*  YB    = (float*) (ws + alloc((size_t)2 * CC * LL * 4));        // (B,C,H,W)
  float*  STATS = (float*) (ws + alloc((size_t)2 * CC * 4));
  bf16_t* XMB   = (bf16_t*)(ws + alloc((size_t)2 * LL * CC * 2));        // (B,L,C)
  float*  F     = (float*) (ws + alloc((size_t)2 * CC * LL * 4));
  bf16_t* F2BF  = (bf16_t*)(ws + alloc((size_t)2 * LL * CC * 2));
  bf16_t* WINP  = (bf16_t*)(ws + alloc((size_t)512 * 128 * 2));
  bf16_t* WXP   = (bf16_t*)(ws + alloc((size_t)48 * 256 * 2));           // padded 40->48
  bf16_t* WOUT  = (bf16_t*)(ws + alloc((size_t)128 * 256 * 2));
  bf16_t* WP1   = (bf16_t*)(ws + alloc((size_t)512 * 512 * 2));
  bf16_t* WP2   = (bf16_t*)(ws + alloc((size_t)512 * 512 * 2));

  float* out_mamba = (float*)d_out;
  float* out_ffn   = out_mamba + (size_t)2 * CC * LL;

  const dim3 blk(256);
  auto cdiv = [](int a, int b) { return (a + b - 1) / b; };

  // weight conversions to bf16 (x_proj padded to 48 rows)
  k_convert_bf16<<<cdiv(512 * 128, 256), blk, 0, stream>>>(in_proj_w, WINP, 512, 128, 512);
  k_convert_bf16<<<cdiv(48 * 256, 256),  blk, 0, stream>>>(x_proj_w,  WXP,  40, 256, 48);
  k_convert_bf16<<<cdiv(128 * 256, 256), blk, 0, stream>>>(out_proj_w,WOUT, 128, 256, 128);
  k_convert_bf16<<<cdiv(512 * 512, 256), blk, 0, stream>>>(pro1_w,    WP1,  512, 512, 512);
  k_convert_bf16<<<cdiv(512 * 512, 256), blk, 0, stream>>>(pro2_w,    WP2,  512, 512, 512);

  k_layernorm<<<2 * LL, blk, 0, stream>>>(x, ln_g, ln_b, XN, XNBF);
  k_gemm_in_proj<<<dim3(MROWS / 16 / 8, 512 / 16), blk, 0, stream>>>(XNBF, WINP, XZ);
  k_conv_silu<<<cdiv(MROWS * DI, 256), blk, 0, stream>>>(XZ, conv_w, conv_b, U, UBF);
  k_gemm_x_proj<<<dim3(MROWS / 16 / 8, 3), blk, 0, stream>>>(UBF, WXP, XDBL);
  k_dt<<<cdiv(MROWS * DI, 256), blk, 0, stream>>>(XDBL, dt_w, dt_b, DT);
  k_scan<<<cdiv(NB * DI * DSS, 256), blk, 0, stream>>>(U, DT, XDBL, A_log, Dp, Y);
  k_gate<<<cdiv(MROWS * DI, 256), blk, 0, stream>>>(XZ, Y, UBF /* now YBF */);
  k_gemm_out_proj<<<dim3(MROWS / 16 / 8, 128 / 16), blk, 0, stream>>>(UBF, WOUT, XN, YB);
  k_bn_stats<<<CC, blk, 0, stream>>>(YB, STATS);
  k_bn_apply<<<cdiv(2 * CC * LL, 256), blk, 0, stream>>>(YB, STATS, bn_g, bn_b, out_mamba, XMB);
  k_gemm_pro1<<<dim3(2 * LL / 16 / 8, 512 / 16), blk, 0, stream>>>(XMB, WP1, pro1_b, F);
  k_dwconv_gelu<<<cdiv(2 * CC * LL, 256), blk, 0, stream>>>(F, dw_w, dw_b, F2BF);
  k_gemm_pro2<<<dim3(2 * LL / 16 / 8, 512 / 16), blk, 0, stream>>>(F2BF, WP2, pro2_b, out_ffn);
}
